// AttentionLayer_15539191677005
// MI455X (gfx1250) — matrix operand order repaired and checked by reading, not compiled, not run on hardware
//
#include <hip/hip_runtime.h>
#include <hip/hip_bf16.h>
#include <stdint.h>

// ---------------- CDNA5 WMMA types ----------------
typedef __attribute__((ext_vector_type(16))) __bf16    v16bf;
typedef __attribute__((ext_vector_type(8)))  float     v8f;
typedef __attribute__((ext_vector_type(4)))  uint32_t  v4u;

struct Frag32B { v4u lo, hi; };
static_assert(sizeof(Frag32B) == 32, "frag must be 32B");

#define BATCH 1024
#define IDIM  256
#define HDIM  64
#define NCOL  (IDIM * HDIM)   // 16384
#define NIT   200
#define MROWS 208             // 200 padded to 13 WMMA m-tiles
#define LDI   264             // item LDS row stride (bf16): 528B, 16B-aligned, bank-skewed
#define LDP   264             // transposed-P LDS row stride (bf16)

// ---------------- helpers ----------------
// round-to-nearest f32->bf16 pair pack: 2x v_add + 1x v_perm_b32
__device__ __forceinline__ uint32_t pack2bf(float a, float b) {
  const uint32_t ua = __builtin_bit_cast(uint32_t, a) + 0x8000u;
  const uint32_t ub = __builtin_bit_cast(uint32_t, b) + 0x8000u;
  return __builtin_amdgcn_perm(ub, ua, 0x07060302u);  // {ub[3:2], ua[3:2]}
}
__device__ __forceinline__ unsigned short bf16_of(float x) {
  return (unsigned short)((__builtin_bit_cast(uint32_t, x) + 0x8000u) >> 16);
}
__device__ __forceinline__ float bf16_to_f32(unsigned short h) {
  return __builtin_bit_cast(float, (uint32_t)h << 16);
}
__device__ __forceinline__ float fast_tanh(float x) {
  x = fminf(fmaxf(x, -15.f), 15.f);
  float e2 = __builtin_amdgcn_exp2f(x * 2.8853900817779268f);   // exp(2x)
  return 1.f - 2.f * __builtin_amdgcn_rcpf(e2 + 1.f);           // v_rcp_f32, no div chain
}
__device__ __forceinline__ float fast_exp(float x) {
  return __builtin_amdgcn_exp2f(x * 1.4426950408889634f);
}
__device__ __forceinline__ v8f wmma_bf16(v16bf a, v16bf b, v8f c) {
  return __builtin_amdgcn_wmma_f32_16x16x32_bf16(false, a, false, b,
                                                 (short)0, c, false, false);
}

// =====================================================================
// Prep A: q (1024x256 f32) -> qbf (bf16)
// =====================================================================
__global__ __launch_bounds__(256) void cvt_q(
    const float* __restrict__ q, unsigned short* __restrict__ qbf) {
  const int i = (blockIdx.x * 256 + threadIdx.x) * 4;       // 256 blocks cover 262144
  const float4 f = *(const float4*)(q + i);
  uint32_t* d = (uint32_t*)(qbf + i);
  d[0] = pack2bf(f.x, f.y);
  d[1] = pack2bf(f.z, f.w);
}

// =====================================================================
// Prep B: Wq (256x16384 f32, [k][n]) -> WqT (16384x256 bf16, [n][k])
// 64x64 tiles through LDS; coalesced f32 reads, b128 bf16 writes.
// =====================================================================
__global__ __launch_bounds__(256) void cvt_transpose_wq(
    const float* __restrict__ Wq, unsigned short* __restrict__ WqT) {
  __shared__ unsigned short tile[64][72];      // [n][k], 144B rows: 16B aligned
  const int k0 = blockIdx.x * 64;              // 4 k-tiles
  const int n0 = blockIdx.y * 64;              // 256 n-tiles
  const int t  = threadIdx.x;

  const int rk = t >> 4;                       // k row group
  const int nn = (t & 15) * 4;                 // n quad
#pragma unroll
  for (int j = 0; j < 4; ++j) {
    const int k = rk + 16 * j;
    const float4 f = *(const float4*)(Wq + (size_t)(k0 + k) * NCOL + n0 + nn);
    tile[nn + 0][k] = bf16_of(f.x);
    tile[nn + 1][k] = bf16_of(f.y);
    tile[nn + 2][k] = bf16_of(f.z);
    tile[nn + 3][k] = bf16_of(f.w);
  }
  __syncthreads();
  const int n = t >> 2, part = t & 3;          // 32B of one n-row per thread
  const unsigned short* s = &tile[n][part * 16];
  v4u a  = *(const v4u*)s;
  v4u b2 = *(const v4u*)(s + 8);
  unsigned short* dst = WqT + (size_t)(n0 + n) * IDIM + k0 + part * 16;
  ((v4u*)dst)[0] = a;
  ((v4u*)dst)[1] = b2;
}

// =====================================================================
// Kernel 1: P = tanh(q @ Wq + bq), stored bf16, (1024 x 16384)
// Block = 256 thr (8 waves), tile 32(M) x 128(N), K-step 32, K=256.
// Sources pre-converted bf16 (qbf, WqT) -> staging is pure vector copies.
// Each wave owns a 16x32 tile (A-fragment reused across 2 WMMAs).
// Double-buffered LDS: 1 barrier/step, global loads overlap WMMA.
// =====================================================================
__global__ __launch_bounds__(256) void proj_tanh_gemm(
    const unsigned short* __restrict__ qbf,   // (1024,256) bf16
    const unsigned short* __restrict__ WqT,   // (16384,256) bf16 [n][k]
    const float* __restrict__ bq,             // (16384)
    unsigned short* __restrict__ P)           // (1024,16384) bf16
{
  constexpr int LDA = 40;                         // 80B rows: 16B aligned
  constexpr int LDB = 40;
  __shared__ unsigned short ldsA[2][32 * LDA];    // M x K  (bf16)
  __shared__ unsigned short ldsBT[2][128 * LDB];  // N x K  (bf16)

  const int tid  = threadIdx.x;
  const int lane = tid & 31;
  const int wave = tid >> 5;
  const int wm = wave >> 2, wn = wave & 3;        // 2x4 waves, 16x32 tile each
  const int bm0 = blockIdx.x * 32;
  const int n0  = blockIdx.y * 128;

  const int r  = lane & 15;                       // M row (A) / N col (B)
  const int hf = lane >> 4;                       // K-half selector

  const int arow = tid >> 3;                      // A copier: row 0..31
  const int ach  = tid & 7;                       //           8B chunk
  const int brow = tid >> 1;                      // B copier: n 0..127
  const int bh   = tid & 1;                       //           16-elem half

  v8f c0 = {0.f, 0.f, 0.f, 0.f, 0.f, 0.f, 0.f, 0.f};
  v8f c1 = {0.f, 0.f, 0.f, 0.f, 0.f, 0.f, 0.f, 0.f};

  uint2 qa;                                       // staged A (4 bf16)
  v4u   wb0, wb1;                                 // staged B (16 bf16)

  auto gload = [&](int ks) {
    const int k0 = ks * 32;
    qa = *(const uint2*)(qbf + (bm0 + arow) * IDIM + k0 + ach * 4);
    const unsigned short* s = WqT + (size_t)(n0 + brow) * IDIM + k0 + bh * 16;
    wb0 = *(const v4u*)s;
    wb1 = *(const v4u*)(s + 8);
  };
  auto lstore = [&](int p) {
    *(uint2*)(&ldsA[p][arow * LDA + ach * 4]) = qa;
    v4u* d = (v4u*)(&ldsBT[p][brow * LDB + bh * 16]);
    d[0] = wb0;
    d[1] = wb1;
  };

  gload(0);
  lstore(0);

  for (int ks = 0; ks < 8; ++ks) {
    const int p = ks & 1;
    __syncthreads();                              // buf[p] ready / buf[p^1] drained
    if (ks < 7) {
      gload(ks + 1);                              // overlap VMEM with WMMA below
      if (ks < 6)                                 // speculative prefetch 2 steps out
        __builtin_prefetch(WqT + (size_t)(n0 + brow) * IDIM + ks * 32 + 64, 0, 0);
    }
    // ---- fragments per ISA layout, ds_load_b128 each ----
    const unsigned short* ap = &ldsA[p][(wm * 16 + r) * LDA + hf * 8];
    Frag32B afr = { *(const v4u*)ap, *(const v4u*)(ap + 16) };
    const unsigned short* bp0 = &ldsBT[p][(wn * 32 + r) * LDB + hf * 8];
    Frag32B bfr0 = { *(const v4u*)bp0, *(const v4u*)(bp0 + 16) };
    const unsigned short* bp1 = &ldsBT[p][(wn * 32 + 16 + r) * LDB + hf * 8];
    Frag32B bfr1 = { *(const v4u*)bp1, *(const v4u*)(bp1 + 16) };

    const v16bf av = __builtin_bit_cast(v16bf, afr);
    c0 = wmma_bf16(av, __builtin_bit_cast(v16bf, bfr0), c0);
    c1 = wmma_bf16(av, __builtin_bit_cast(v16bf, bfr1), c1);

    if (ks < 7) lstore(p ^ 1);                    // fill other buffer
  }

  // ---- fused bias + tanh, store bf16 ----
  const int ng0 = n0 + wn * 32 + r;
  const int ng1 = ng0 + 16;
  const float bias0 = bq[ng0];
  const float bias1 = bq[ng1];
#pragma unroll
  for (int v = 0; v < 8; ++v) {
    const int mg = bm0 + wm * 16 + v + 8 * hf;    // C layout: M = vgpr + 8*half
    P[(size_t)mg * NCOL + ng0] = bf16_of(fast_tanh(c0[v] + bias0));
    P[(size_t)mg * NCOL + ng1] = bf16_of(fast_tanh(c1[v] + bias1));
  }
}

// =====================================================================
// Kernel 2: per-batch fused attention. 1 workgroup (256 thr) per batch.
// item tile read once -> LDS bf16; dotted via WMMA folded into score;
// ZAM weights; weighted sum of items from LDS. ~145 KB LDS -> 2 WG/WGP.
// =====================================================================
__global__ __launch_bounds__(256) void attn_fused(
    const float* __restrict__ item,           // (1024,200,256)
    const unsigned short* __restrict__ P,     // (1024,16384) bf16
    const float* __restrict__ Wr,             // (64)
    float* __restrict__ out)                  // (1024,256)
{
  extern __shared__ unsigned char smem[];
  unsigned short* itemS = (unsigned short*)smem;        // MROWS x LDI
  unsigned short* ptS   = itemS + MROWS * LDI;          // 64 x LDP  (P^T)
  float* scoreS = (float*)(ptS + 64 * LDP);             // 256
  float* wsum   = scoreS + 256;                         // 8

  const int b    = blockIdx.x;
  const int tid  = threadIdx.x;
  const int lane = tid & 31;
  const int wave = tid >> 5;

  scoreS[tid] = 0.f;

  // ---- item block -> LDS bf16 (single global pass over 210 MB total) ----
  const float* ibase = item + (size_t)b * NIT * IDIM;
  for (int idx = tid; idx < NIT * (IDIM / 4); idx += 256) {
    const int row = idx >> 6, c4 = idx & 63;
    __builtin_prefetch(ibase + row * IDIM + c4 * 4 + 1024, 0, 0);  // ~1 iter ahead
    const float4 f = *(const float4*)(ibase + row * IDIM + c4 * 4);
    uint32_t* d = (uint32_t*)(itemS + row * LDI + c4 * 4);
    d[0] = pack2bf(f.x, f.y);
    d[1] = pack2bf(f.z, f.w);
  }
  for (int idx = tid; idx < (MROWS - NIT) * (IDIM / 4); idx += 256) {
    const int row = NIT + (idx >> 6), c4 = idx & 63;
    uint32_t* d = (uint32_t*)(itemS + row * LDI + c4 * 4);
    d[0] = 0; d[1] = 0;                       // zero pad rows -> score contrib 0
  }
  // ---- P tile (256 x 64) -> LDS transposed ptS[h][i] ----
  {
    const unsigned short* pb = P + (size_t)b * NCOL;
    const int h  = tid & 63;
    const int i0 = (tid >> 6) * 2;
#pragma unroll 4
    for (int j = 0; j < 32; ++j) {
      const int i = i0 + 8 * j;
      const uint32_t lo = pb[(i    ) * HDIM + h];
      const uint32_t hi = pb[(i + 1) * HDIM + h];
      *(uint32_t*)(ptS + h * LDP + i) = lo | (hi << 16);
    }
  }
  __syncthreads();

  // ---- dotted(208x64) = item @ P^T via WMMA; fold dot(., Wr) into score ----
  const int r  = lane & 15;
  const int hf = lane >> 4;
  for (int tile = wave; tile < 52; tile += 8) {     // 13 m-tiles x 4 h-tiles
    const int m0 = (tile >> 2) * 16, h0 = (tile & 3) * 16;
    v8f c = {0.f, 0.f, 0.f, 0.f, 0.f, 0.f, 0.f, 0.f};
#pragma unroll
    for (int ks = 0; ks < 8; ++ks) {
      const int k0 = ks * 32;
      const unsigned short* ap = itemS + (m0 + r) * LDI + k0 + hf * 8;
      Frag32B afr = { *(const v4u*)ap, *(const v4u*)(ap + 16) };
      const unsigned short* bp = ptS + (h0 + r) * LDP + k0 + hf * 8;
      Frag32B bfr = { *(const v4u*)bp, *(const v4u*)(bp + 16) };
      c = wmma_bf16(__builtin_bit_cast(v16bf, afr),
                    __builtin_bit_cast(v16bf, bfr), c);
    }
    const float wr = Wr[h0 + r];                    // lane's N column weight
#pragma unroll
    for (int v = 0; v < 8; ++v) {                   // row m0 + v + 8*hf
      float pp = c[v] * wr;
      pp += __shfl_xor(pp, 1);
      pp += __shfl_xor(pp, 2);
      pp += __shfl_xor(pp, 4);
      pp += __shfl_xor(pp, 8);                      // sum 16 N-lanes
      if (r == 0) atomicAdd(&scoreS[m0 + v + 8 * hf], pp);   // ds_add_f32
    }
  }
  __syncthreads();

  // ---- ZAM: weight = exp(score) / (1 + sum exp(score)) ----
  float es = 0.f;
  if (tid < NIT) es = fast_exp(fminf(scoreS[tid], 30.f));
  float s = es;
#pragma unroll
  for (int m = 1; m < 32; m <<= 1) s += __shfl_xor(s, m);
  if (lane == 0) wsum[wave] = s;
  __syncthreads();
  float tot = 0.f;
#pragma unroll
  for (int ww = 0; ww < 8; ++ww) tot += wsum[ww];
  const float wgt = es * __builtin_amdgcn_rcpf(1.f + tot);   // no v_div chain
  scoreS[tid] = wgt;                                // own-slot only: race-free
  __syncthreads();

  // ---- entity[i] = sum_m w[m] * item[m][i]  (items from LDS) ----
  float acc = 0.f;
#pragma unroll 4
  for (int m = 0; m < NIT; ++m)
    acc = fmaf(scoreS[m], bf16_to_f32(itemS[m * LDI + tid]), acc);
  out[(size_t)b * IDIM + tid] = acc;
}

// =====================================================================
extern "C" void kernel_launch(void* const* d_in, const int* in_sizes, int n_in,
                              void* d_out, int out_size, void* d_ws, size_t ws_size,
                              hipStream_t stream) {
  const float* item = (const float*)d_in[0];   // (1024,200,256)
  const float* q    = (const float*)d_in[1];   // (1024,256)
  const float* Wq   = (const float*)d_in[2];   // (256,16384)
  const float* bq   = (const float*)d_in[3];   // (16384)
  const float* Wr   = (const float*)d_in[4];   // (64)
  float* out = (float*)d_out;                  // (1024,256)

  // workspace carve: P | WqT | qbf  (bf16)  ~42.5 MB total
  unsigned short* P   = (unsigned short*)d_ws;             // 1024*16384
  unsigned short* WqT = P   + (size_t)BATCH * NCOL;        // 16384*256
  unsigned short* qbf = WqT + (size_t)NCOL * IDIM;         // 1024*256

  cvt_q<<<(BATCH * IDIM) / (256 * 4), 256, 0, stream>>>(q, qbf);
  dim3 gT(IDIM / 64, NCOL / 64);               // 4 x 256
  cvt_transpose_wq<<<gT, 256, 0, stream>>>(Wq, WqT);

  dim3 g1(BATCH / 32, NCOL / 128);             // 32 x 128 blocks
  proj_tanh_gemm<<<g1, 256, 0, stream>>>(qbf, WqT, bq, P);

  const size_t smem = (size_t)(MROWS * LDI + 64 * LDP) * sizeof(unsigned short)
                    + (256 + 8) * sizeof(float);          // ~145 KB (< 320 KB/WGP)
  attn_fused<<<BATCH, 256, smem, stream>>>(item, P, Wr, out);
}
